// CausalTransformerBlock_71906342469951
// MI455X (gfx1250) — matrix operand order repaired and checked by reading, not compiled, not run on hardware
//
#include <hip/hip_runtime.h>
#include <hip/hip_bf16.h>
#include <math.h>

// ---------------------------------------------------------------------------
// CausalTransformerBlock for MI455X (gfx1250, wave32, WMMA 16x16x32 f16->f32)
// B=4 S=4096 H=1024 NH=1 HD=1024 D_FF=4096 R=2 k_top=409
// Round 3: fix async-copy builtin signature (AS1/AS3 int4* operands).
// ---------------------------------------------------------------------------

typedef _Float16 f16;
typedef _Float16 v16h __attribute__((ext_vector_type(16)));
typedef _Float16 v8h  __attribute__((ext_vector_type(8)));
typedef float    v8f  __attribute__((ext_vector_type(8)));
typedef int      v4i  __attribute__((ext_vector_type(4)));

#define BDIM 1024   // hidden
#define DFF  4096
#define SEQ  4096
#define NB   4
#define NTOK (NB*SEQ)          // 16384
#define KTOP 409

#define BM 128
#define BN 128
#define BK 32
#define BKP 48                  // padded LDS stride (halfs): 96B rows, 16B aligned chunks

#define SHUF16(lo,hi) __builtin_shufflevector(lo,hi,0,1,2,3,4,5,6,7,8,9,10,11,12,13,14,15)

#define AS1 __attribute__((address_space(1)))
#define AS3 __attribute__((address_space(3)))

#if defined(__HIP_DEVICE_COMPILE__) && __has_builtin(__builtin_amdgcn_global_load_async_to_lds_b128)
#define USE_ASYNC_CP 1
#else
#define USE_ASYNC_CP 0
#endif

// 16-byte global -> LDS copy: async (no VGPR roundtrip, ASYNCcnt) when available
__device__ __forceinline__ void cp16_g2l(const f16* __restrict__ g, f16* __restrict__ l) {
#if USE_ASYNC_CP
    __builtin_amdgcn_global_load_async_to_lds_b128((AS1 v4i*)g, (AS3 v4i*)l, 0, 0);
#else
    *(v8h*)l = *(const v8h*)g;
#endif
}

template <int N>
__device__ __forceinline__ void wait_async() {
#if defined(__gfx1250__)
#if USE_ASYNC_CP
    asm volatile("s_wait_asynccnt %0" :: "n"(N) : "memory");
#endif
#endif
}

// ---------------- shared WMMA main loop: C(128x128) += A(128xK) * W(128xK)^T
// A row-major [M,K] f16, W row-major [N,K] f16 (torch Linear weight layout).
// Per wave32: 4(M) x 2(N) accumulator tiles of 16x16. Double-buffered LDS.
__device__ __forceinline__ void wmma_mainloop(const f16* __restrict__ A, int lda,
                                              const f16* __restrict__ W, int ldw,
                                              int K, v8f (&acc)[4][2]) {
    __shared__ f16 As[2][BM * BKP];
    __shared__ f16 Ws[2][BN * BKP];
    const int t      = threadIdx.x;
    const int lane   = t & 31;
    const int wid    = t >> 5;
    const int wm     = wid & 1;       // 2 waves over M
    const int wn     = wid >> 1;      // 4 waves over N
    const int lr     = t >> 1;        // load row 0..127
    const int lc     = (t & 1) << 4;  // load col 0 or 16 (halfs)
    const int laneHi = lane >> 4;
    const int lm     = lane & 15;

    const f16* agBase = A + (long long)lr * lda + lc;
    const f16* wgBase = W + (long long)lr * ldw + lc;
    const int  lo     = lr * BKP + lc;

    auto issue = [&](int kk, int buf) {
        cp16_g2l(agBase + kk,     &As[buf][lo]);
        cp16_g2l(agBase + kk + 8, &As[buf][lo + 8]);
        cp16_g2l(wgBase + kk,     &Ws[buf][lo]);
        cp16_g2l(wgBase + kk + 8, &Ws[buf][lo + 8]);
    };

    const int nk = K / BK;
    issue(0, 0);
    for (int it = 0; it < nk; ++it) {
        const int buf = it & 1;
        if (it + 1 < nk) {
            issue((it + 1) * BK, buf ^ 1);   // prefetch next slab into other buffer
            wait_async<4>();                 // in-order completion: current slab landed
        } else {
            wait_async<0>();
        }
        __syncthreads();

        v16h af[4], wf[2];
#pragma unroll
        for (int j = 0; j < 2; ++j) {
            // B-operand: lane holds W row (= B column) n=lm, contiguous K chunk 0/16
            const f16* p = &Ws[buf][(wn * 32 + j * 16 + lm) * BKP + laneHi * 16];
            v8h l0 = *(const v8h*)p;
            v8h h0 = *(const v8h*)(p + 8);
            wf[j] = SHUF16(l0, h0);
        }
#pragma unroll
        for (int i = 0; i < 4; ++i) {
            // A-operand: lanes<16 hold K 0..15; lanes>=16 hold 16..31
            const f16* p = &As[buf][(wm * 64 + i * 16 + lm) * BKP + laneHi * 16];
            v8h l0 = *(const v8h*)p;
            v8h h0 = *(const v8h*)(p + 8);
            af[i] = SHUF16(l0, h0);
        }
#pragma unroll
        for (int i = 0; i < 4; ++i)
#pragma unroll
            for (int j = 0; j < 2; ++j)
                acc[i][j] = __builtin_amdgcn_wmma_f32_16x16x32_f16(
                    false, af[i], false, wf[j], (short)0, acc[i][j], false, false);
        __syncthreads();
    }
}

// epilogue coordinate helpers (16x16 f32 C layout: VGPR e -> M=e+8*(lane>=16), N=lane&15)
#define EPI_PROLOG                                   \
    const int lane = threadIdx.x & 31;               \
    const int wid  = threadIdx.x >> 5;               \
    const int wm   = wid & 1, wn = wid >> 1;         \
    const int laneHi = lane >> 4, lm = lane & 15;

// ---------------- GEMM: out_f16 = A*W^T (+bias), batched via blockIdx.z ------
__global__ __launch_bounds__(256) void gemm_f16_out(
        const f16* __restrict__ A, long long sA, int lda,
        const f16* __restrict__ W, long long sW, int K,
        const float* __restrict__ bias,
        f16* __restrict__ out, long long sO, int N) {
    const int r0 = blockIdx.y * BM, c0 = blockIdx.x * BN;
    const f16* Ab = A + (long long)blockIdx.z * sA + (long long)r0 * lda;
    const f16* Wb = W + (long long)blockIdx.z * sW + (long long)c0 * K;
    f16* Ob = out + (long long)blockIdx.z * sO;
    v8f acc[4][2] = {};
    wmma_mainloop(Ab, lda, Wb, K, K, acc);
    EPI_PROLOG
#pragma unroll
    for (int i = 0; i < 4; ++i)
#pragma unroll
        for (int j = 0; j < 2; ++j) {
            int col = c0 + wn * 32 + j * 16 + lm;
            float bv = bias ? bias[col] : 0.0f;
#pragma unroll
            for (int e = 0; e < 8; ++e) {
                int row = r0 + wm * 64 + i * 16 + e + 8 * laneHi;
                Ob[(long long)row * N + col] = (f16)(acc[i][j][e] + bv);
            }
        }
}

// ---------------- V projection, stored transposed: vT[b][d][s] ---------------
__global__ __launch_bounds__(256) void gemm_vT(
        const f16* __restrict__ A, int lda,
        const f16* __restrict__ W, int K,
        const float* __restrict__ bias,
        f16* __restrict__ vT) {
    const int r0 = blockIdx.y * BM, c0 = blockIdx.x * BN;
    v8f acc[4][2] = {};
    wmma_mainloop(A + (long long)r0 * lda, lda, W + (long long)c0 * K, K, K, acc);
    EPI_PROLOG
#pragma unroll
    for (int i = 0; i < 4; ++i)
#pragma unroll
        for (int j = 0; j < 2; ++j) {
            int col = c0 + wn * 32 + j * 16 + lm;
            float bv = bias[col];
#pragma unroll
            for (int e = 0; e < 8; ++e) {
                int row = r0 + wm * 64 + i * 16 + e + 8 * laneHi;   // global token
                int b = row >> 12, s = row & (SEQ - 1);
                vT[((long long)b * BDIM + col) * SEQ + s] = (f16)(acc[i][j][e] + bv);
            }
        }
}

// ---------------- scores = Q K^T / 32, causal+key mask, rank-2 bias ----------
__global__ __launch_bounds__(256) void gemm_scores(
        const f16* __restrict__ Q, const f16* __restrict__ Km,
        const float* __restrict__ aQ,   // [2][NTOK] strength_r*(h1.cause_r + cb_r)
        const float* __restrict__ eK,   // [2][NTOK] h1.eff_w_r
        const float* __restrict__ eB,   // [NTOK]    h1.(eff_b0+eff_b1)
        const unsigned char* __restrict__ amask,
        float* __restrict__ scores) {
    const int b  = blockIdx.z;
    const int r0 = blockIdx.y * BM, c0 = blockIdx.x * BN;
    float* Ob = scores + (long long)b * SEQ * SEQ;
    const int tokB = b * SEQ;

    if (c0 > r0 + (BM - 1)) {       // tile fully above diagonal: epilogue only
        for (int idx = threadIdx.x; idx < BM * BN; idx += 256) {
            int r = idx >> 7, c = idx & 127;
            int row = r0 + r, col = c0 + c;
            float bias = aQ[tokB + row] * eK[tokB + col]
                       + aQ[NTOK + tokB + row] * eK[NTOK + tokB + col]
                       + eB[tokB + col];
            Ob[(long long)row * SEQ + col] = -10000.0f + bias;
        }
        return;
    }
    v8f acc[4][2] = {};
    wmma_mainloop(Q + (long long)b * SEQ * BDIM + (long long)r0 * BDIM, BDIM,
                  Km + (long long)b * SEQ * BDIM + (long long)c0 * BDIM, BDIM,
                  BDIM, acc);
    EPI_PROLOG
#pragma unroll
    for (int i = 0; i < 4; ++i)
#pragma unroll
        for (int j = 0; j < 2; ++j) {
            int col = c0 + wn * 32 + j * 16 + lm;
            float ek0 = eK[tokB + col], ek1 = eK[NTOK + tokB + col], eb = eB[tokB + col];
            bool keyok = amask[tokB + col] != 0;
#pragma unroll
            for (int e = 0; e < 8; ++e) {
                int row = r0 + wm * 64 + i * 16 + e + 8 * laneHi;
                float v = acc[i][j][e] * 0.03125f;              // 1/sqrt(1024)
                if (col > row || !keyok) v = -10000.0f;         // mask BEFORE bias (matches ref)
                v += aQ[tokB + row] * ek0 + aQ[NTOK + tokB + row] * ek1 + eb;
                Ob[(long long)row * SEQ + col] = v;
            }
        }
}

// ---------------- ctx-proj / fc2: out_f32 = A*W^T + bias + residual ----------
__global__ __launch_bounds__(256) void gemm_f32_res(
        const f16* __restrict__ A, int lda,
        const f16* __restrict__ W, int K,
        const float* __restrict__ bias,
        const float* __restrict__ resid,
        float* __restrict__ out, int N) {
    const int r0 = blockIdx.y * BM, c0 = blockIdx.x * BN;
    v8f acc[4][2] = {};
    wmma_mainloop(A + (long long)r0 * lda, lda, W + (long long)c0 * K, K, K, acc);
    EPI_PROLOG
#pragma unroll
    for (int i = 0; i < 4; ++i)
#pragma unroll
        for (int j = 0; j < 2; ++j) {
            int col = c0 + wn * 32 + j * 16 + lm;
            float bv = bias[col];
#pragma unroll
            for (int e = 0; e < 8; ++e) {
                int row = r0 + wm * 64 + i * 16 + e + 8 * laneHi;
                long long o = (long long)row * N + col;
                out[o] = acc[i][j][e] + bv + resid[o];
            }
        }
}

// ---------------- fc1: out_f16 = gelu(A*W^T + bias) --------------------------
__global__ __launch_bounds__(256) void gemm_gelu(
        const f16* __restrict__ A, int lda,
        const f16* __restrict__ W, int K,
        const float* __restrict__ bias,
        f16* __restrict__ out, int N) {
    const int r0 = blockIdx.y * BM, c0 = blockIdx.x * BN;
    v8f acc[4][2] = {};
    wmma_mainloop(A + (long long)r0 * lda, lda, W + (long long)c0 * K, K, K, acc);
    EPI_PROLOG
#pragma unroll
    for (int i = 0; i < 4; ++i)
#pragma unroll
        for (int j = 0; j < 2; ++j) {
            int col = c0 + wn * 32 + j * 16 + lm;
            float bv = bias[col];
#pragma unroll
            for (int e = 0; e < 8; ++e) {
                int row = r0 + wm * 64 + i * 16 + e + 8 * laneHi;
                float v = acc[i][j][e] + bv;
                v = 0.5f * v * (1.0f + erff(v * 0.70710678118654752f));   // exact gelu
                out[(long long)row * N + col] = (f16)v;
            }
        }
}

// ---------------- block reductions -------------------------------------------
__device__ __forceinline__ float blk_sum(float v, float* s) {
    int t = threadIdx.x;
    s[t] = v; __syncthreads();
    for (int k = 128; k > 0; k >>= 1) { if (t < k) s[t] += s[t + k]; __syncthreads(); }
    float r = s[0]; __syncthreads();
    return r;
}
__device__ __forceinline__ float blk_max(float v, float* s) {
    int t = threadIdx.x;
    s[t] = v; __syncthreads();
    for (int k = 128; k > 0; k >>= 1) { if (t < k) s[t] = fmaxf(s[t], s[t + k]); __syncthreads(); }
    float r = s[0]; __syncthreads();
    return r;
}

// ---------------- LayerNorm (+ per-token causal-bias scalars for LN1) --------
template <bool DOTS>
__global__ __launch_bounds__(256) void ln_kernel(
        const float* __restrict__ x, const float* __restrict__ g, const float* __restrict__ b,
        f16* __restrict__ h16,
        const float* __restrict__ cause_w, const float* __restrict__ cause_b,
        const float* __restrict__ eff_w,   const float* __restrict__ eff_b,
        const float* __restrict__ strengths,
        float* __restrict__ aQ, float* __restrict__ eK, float* __restrict__ eB) {
    __shared__ float sbuf[256];
    const int tok = blockIdx.x, t = threadIdx.x;
    const float* xr = x + (long long)tok * BDIM;
    float xv[4];
#pragma unroll
    for (int i = 0; i < 4; ++i) xv[i] = xr[t * 4 + i];
    float sum = xv[0] + xv[1] + xv[2] + xv[3];
    float sq  = xv[0]*xv[0] + xv[1]*xv[1] + xv[2]*xv[2] + xv[3]*xv[3];
    float mean = blk_sum(sum, sbuf) * (1.0f / BDIM);
    float var  = blk_sum(sq,  sbuf) * (1.0f / BDIM) - mean * mean;
    float rstd = rsqrtf(var + 1e-5f);
    float d[6] = {0,0,0,0,0,0};
#pragma unroll
    for (int i = 0; i < 4; ++i) {
        int j = t * 4 + i;
        float h = (xv[i] - mean) * rstd * g[j] + b[j];
        h16[(long long)tok * BDIM + j] = (f16)h;
        if (DOTS) {
            d[0] += h * cause_w[j];           d[1] += h * cause_w[BDIM + j];
            d[2] += h * eff_w[j];             d[3] += h * eff_w[BDIM + j];
            d[4] += h * eff_b[j];             d[5] += h * eff_b[BDIM + j];
        }
    }
    if (DOTS) {
#pragma unroll
        for (int i = 0; i < 6; ++i) d[i] = blk_sum(d[i], sbuf);
        if (t == 0) {
            aQ[tok]        = strengths[0] * (d[0] + cause_b[0]);
            aQ[NTOK + tok] = strengths[1] * (d[1] + cause_b[1]);
            eK[tok] = d[2]; eK[NTOK + tok] = d[3];
            eB[tok] = d[4] + d[5];
        }
    }
}

// ---------------- row softmax over 4096 keys, f32 in-place + f16 copy --------
__global__ __launch_bounds__(256) void softmax_kernel(float* __restrict__ scores,
                                                      f16* __restrict__ attn16) {
    __shared__ float sbuf[256];
    const long long row = blockIdx.x;           // token-major == [B,1,S,S] contiguous
    float* p = scores + row * SEQ;
    f16*   q = attn16 + row * SEQ;
    const int t = threadIdx.x;
    float m = -3.4e38f;
    for (int i = t; i < SEQ; i += 256) m = fmaxf(m, p[i]);
    m = blk_max(m, sbuf);
    float s = 0.0f;
    for (int i = t; i < SEQ; i += 256) s += expf(p[i] - m);
    s = blk_sum(s, sbuf);
    float inv = 1.0f / s;
    for (int i = t; i < SEQ; i += 256) {
        float a = expf(p[i] - m) * inv;
        p[i] = a;
        q[i] = (f16)a;
    }
}

// ---------------- top-k(409) radix select on f16 bit-keys; zero ff below -----
__global__ __launch_bounds__(256) void topk_mask_kernel(const f16* __restrict__ logits,
                                                        f16* __restrict__ ff) {
    __shared__ int hist[256];
    __shared__ int sHi, sNeed, sThr;
    const long long row = blockIdx.x;
    const unsigned short* lp = (const unsigned short*)(logits + row * DFF);
    f16* fp = ff + row * DFF;
    const int t = threadIdx.x;

    hist[t] = 0; __syncthreads();
    for (int i = t; i < DFF; i += 256) {
        unsigned short u = lp[i];
        unsigned short k = u ^ ((u & 0x8000) ? 0xFFFFu : 0x8000u);   // order-preserving
        atomicAdd(&hist[k >> 8], 1);
    }
    __syncthreads();
    if (t == 0) {
        int cum = 0;
        for (int bin = 255; bin >= 0; --bin) {
            cum += hist[bin];
            if (cum >= KTOP) { sHi = bin; sNeed = KTOP - (cum - hist[bin]); break; }
        }
    }
    __syncthreads();
    hist[t] = 0; __syncthreads();
    const int hi = sHi;
    for (int i = t; i < DFF; i += 256) {
        unsigned short u = lp[i];
        unsigned short k = u ^ ((u & 0x8000) ? 0xFFFFu : 0x8000u);
        if ((k >> 8) == hi) atomicAdd(&hist[k & 255], 1);
    }
    __syncthreads();
    if (t == 0) {
        int cum = 0, need = sNeed;
        for (int bin = 255; bin >= 0; --bin) {
            cum += hist[bin];
            if (cum >= need) { sThr = (hi << 8) | bin; break; }
        }
    }
    __syncthreads();
    const int thr = sThr;
    for (int i = t; i < DFF; i += 256) {
        unsigned short u = lp[i];
        unsigned short k = u ^ ((u & 0x8000) ? 0xFFFFu : 0x8000u);
        if ((int)k < thr) fp[i] = (f16)0.0f;
    }
}

// ---------------- f32 -> f16 weight conversion -------------------------------
__global__ __launch_bounds__(256) void cvt_kernel(const float* __restrict__ in,
                                                  f16* __restrict__ out, int n) {
    int i = (blockIdx.x * 256 + threadIdx.x) * 4;
    if (i + 3 < n) {
#pragma unroll
        for (int j = 0; j < 4; ++j) out[i + j] = (f16)in[i + j];
    }
}

// ---------------------------------------------------------------------------
extern "C" void kernel_launch(void* const* d_in, const int* in_sizes, int n_in,
                              void* d_out, int out_size, void* d_ws, size_t ws_size,
                              hipStream_t stream) {
    const float* x      = (const float*)d_in[0];
    const unsigned char* amask = (const unsigned char*)d_in[1];
    const float* wq = (const float*)d_in[2];  const float* bq = (const float*)d_in[3];
    const float* wk = (const float*)d_in[4];  const float* bk = (const float*)d_in[5];
    const float* wv = (const float*)d_in[6];  const float* bv = (const float*)d_in[7];
    const float* wo = (const float*)d_in[8];  const float* bo = (const float*)d_in[9];
    const float* ln1_g = (const float*)d_in[10]; const float* ln1_b = (const float*)d_in[11];
    const float* ln2_g = (const float*)d_in[12]; const float* ln2_b = (const float*)d_in[13];
    const float* fc1_w = (const float*)d_in[14]; const float* fc1_b = (const float*)d_in[15];
    const float* fc2_w = (const float*)d_in[16]; const float* fc2_b = (const float*)d_in[17];
    const float* rt_w  = (const float*)d_in[18]; const float* rt_b  = (const float*)d_in[19];
    const float* cause_w = (const float*)d_in[20]; const float* cause_b = (const float*)d_in[21];
    const float* eff_w   = (const float*)d_in[22]; const float* eff_b   = (const float*)d_in[23];
    const float* strengths = (const float*)d_in[24];

    float* out_x    = (float*)d_out;                          // [B,S,H]
    float* out_attn = (float*)d_out + (long long)NTOK * BDIM; // [B,1,S,S]

    // ---- workspace bump allocator (all sizes 256B-aligned) ----
    char* w = (char*)d_ws;
    size_t off = 0;
    auto alloc = [&](size_t bytes) { char* p = w + off; off += (bytes + 255) & ~size_t(255); return p; };
    f16* h1    = (f16*)alloc((size_t)NTOK * BDIM * 2);
    f16* q16   = (f16*)alloc((size_t)NTOK * BDIM * 2);
    f16* k16   = (f16*)alloc((size_t)NTOK * BDIM * 2);
    f16* vT    = (f16*)alloc((size_t)NB * BDIM * SEQ * 2);
    f16* ctx16 = (f16*)alloc((size_t)NTOK * BDIM * 2);
    f16* h2    = (f16*)alloc((size_t)NTOK * BDIM * 2);
    float* x2  = (float*)alloc((size_t)NTOK * BDIM * 4);
    f16* ff16  = (f16*)alloc((size_t)NTOK * DFF * 2);
    f16* attn16 = (f16*)alloc((size_t)NB * SEQ * SEQ * 2);   // reused as router logits
    f16* wq16 = (f16*)alloc((size_t)BDIM * BDIM * 2);
    f16* wk16 = (f16*)alloc((size_t)BDIM * BDIM * 2);
    f16* wv16 = (f16*)alloc((size_t)BDIM * BDIM * 2);
    f16* wo16 = (f16*)alloc((size_t)BDIM * BDIM * 2);
    f16* fc1w16 = (f16*)alloc((size_t)DFF * BDIM * 2);
    f16* fc2w16 = (f16*)alloc((size_t)BDIM * DFF * 2);
    f16* rtw16  = (f16*)alloc((size_t)DFF * BDIM * 2);
    float* aQ = (float*)alloc((size_t)2 * NTOK * 4);
    float* eK = (float*)alloc((size_t)2 * NTOK * 4);
    float* eB = (float*)alloc((size_t)NTOK * 4);
    f16* logits16 = attn16;   // attention probs no longer needed in f16 by FFN stage

    dim3 blk(256);
    // weight conversions to f16
    int nw1 = BDIM * BDIM, nw4 = DFF * BDIM;
    cvt_kernel<<<nw1 / 1024, blk, 0, stream>>>(wq, wq16, nw1);
    cvt_kernel<<<nw1 / 1024, blk, 0, stream>>>(wk, wk16, nw1);
    cvt_kernel<<<nw1 / 1024, blk, 0, stream>>>(wv, wv16, nw1);
    cvt_kernel<<<nw1 / 1024, blk, 0, stream>>>(wo, wo16, nw1);
    cvt_kernel<<<nw4 / 1024, blk, 0, stream>>>(fc1_w, fc1w16, nw4);
    cvt_kernel<<<nw4 / 1024, blk, 0, stream>>>(fc2_w, fc2w16, nw4);
    cvt_kernel<<<nw4 / 1024, blk, 0, stream>>>(rt_w,  rtw16,  nw4);

    // LN1 + rank-2 causal-bias scalars
    ln_kernel<true><<<NTOK, blk, 0, stream>>>(x, ln1_g, ln1_b, h1,
                                              cause_w, cause_b, eff_w, eff_b, strengths,
                                              aQ, eK, eB);
    // Q, K projections; V stored transposed
    gemm_f16_out<<<dim3(BDIM / BN, NTOK / BM, 1), blk, 0, stream>>>(
        h1, 0, BDIM, wq16, 0, BDIM, bq, q16, 0, BDIM);
    gemm_f16_out<<<dim3(BDIM / BN, NTOK / BM, 1), blk, 0, stream>>>(
        h1, 0, BDIM, wk16, 0, BDIM, bk, k16, 0, BDIM);
    gemm_vT<<<dim3(BDIM / BN, NTOK / BM, 1), blk, 0, stream>>>(h1, BDIM, wv16, BDIM, bv, vT);

    // scores (+mask +rank-2 bias) -> d_out attn region, then softmax
    gemm_scores<<<dim3(SEQ / BN, SEQ / BM, NB), blk, 0, stream>>>(
        q16, k16, aQ, eK, eB, amask, out_attn);
    softmax_kernel<<<NTOK, blk, 0, stream>>>(out_attn, attn16);

    // ctx = attn @ V  (A = attn f16 [S,S], W = vT [HD,S] per batch)
    gemm_f16_out<<<dim3(BDIM / BN, SEQ / BM, NB), blk, 0, stream>>>(
        attn16, (long long)SEQ * SEQ, SEQ,
        vT, (long long)BDIM * SEQ, SEQ,
        nullptr, ctx16, (long long)SEQ * BDIM, BDIM);

    // out-proj + residual -> x2
    gemm_f32_res<<<dim3(BDIM / BN, NTOK / BM, 1), blk, 0, stream>>>(
        ctx16, BDIM, wo16, BDIM, bo, x, x2, BDIM);

    // LN2
    ln_kernel<false><<<NTOK, blk, 0, stream>>>(x2, ln2_g, ln2_b, h2,
                                               nullptr, nullptr, nullptr, nullptr, nullptr,
                                               nullptr, nullptr, nullptr);
    // fc1 + gelu, router logits
    gemm_gelu<<<dim3(DFF / BN, NTOK / BM, 1), blk, 0, stream>>>(
        h2, BDIM, fc1w16, BDIM, fc1_b, ff16, DFF);
    gemm_f16_out<<<dim3(DFF / BN, NTOK / BM, 1), blk, 0, stream>>>(
        h2, 0, BDIM, rtw16, 0, BDIM, rt_b, logits16, 0, DFF);

    // top-409 per row, zero the rest of ff in place
    topk_mask_kernel<<<NTOK, blk, 0, stream>>>(logits16, ff16);

    // fc2 + residual -> final x output
    gemm_f32_res<<<dim3(BDIM / BN, NTOK / BM, 1), blk, 0, stream>>>(
        ff16, DFF, fc2w16, DFF, fc2_b, x2, out_x, BDIM);
}
